// MSDNContext_50525995270250
// MI455X (gfx1250) — matrix operand order repaired and checked by reading, not compiled, not run on hardware
//
#include <hip/hip_runtime.h>
#include <cstdint>

// ---------------------------------------------------------------------------
// Problem constants (from the reference)
// ---------------------------------------------------------------------------
#define NI_   4096
#define NR_   65536
#define H_    1024
#define POOL_ 2048
#define GW_   128

// ---------------------------------------------------------------------------
// CDNA5 WMMA types
// ---------------------------------------------------------------------------
typedef __attribute__((ext_vector_type(16))) __bf16 v16bf;
typedef __attribute__((ext_vector_type(8)))  float  v8f;

union Frag {
    unsigned int u[8];
    v16bf        v;
};

__device__ __forceinline__ unsigned short f2bf(float f) {
    unsigned int u = __float_as_uint(f);
    unsigned int r = u + 0x7fffu + ((u >> 16) & 1u);   // round-to-nearest-even
    return (unsigned short)(r >> 16);
}

__device__ __forceinline__ float sigmoidf_(float x) {
    return 1.0f / (1.0f + __expf(-x));
}

// A-matrix 16x32 bf16 fragment from row-major LDS tile (ld in ushorts).
// Per ISA 7.12.2: lane l (r=l&15,hi=l>>4), VGPR v holds K = (v>>2)*16 + hi*8 + (v&3)*2.
__device__ __forceinline__ void load_a_frag(const unsigned short* a, int ld, int mbase,
                                            int lane, Frag& f) {
    int r  = lane & 15;
    int hi = lane >> 4;
    const unsigned short* row = a + (size_t)(mbase + r) * ld;
#pragma unroll
    for (int v = 0; v < 8; ++v) {
        int k = ((v >> 2) << 4) + (hi << 3) + ((v & 3) << 1);
        f.u[v] = *(const unsigned int*)(row + k);
    }
}

// B-matrix 32x16 bf16 fragment from LDS tile stored TRANSPOSED: bt[n][k], ld in ushorts.
// Lane l holds column n=l&15; VGPR v -> K = (v>>2)*16 + hi*8 + (v&3)*2 (same as A).
__device__ __forceinline__ void load_b_frag(const unsigned short* bt, int ld, int nbase,
                                            int lane, Frag& f) {
    int n  = lane & 15;
    int hi = lane >> 4;
    const unsigned short* col = bt + (size_t)(nbase + n) * ld;
#pragma unroll
    for (int v = 0; v < 8; ++v) {
        int k = ((v >> 2) << 4) + (hi << 3) + ((v & 3) << 1);
        f.u[v] = *(const unsigned int*)(col + k);
    }
}

__device__ __forceinline__ v8f wmma_bf16(const Frag& a, const Frag& b, v8f c) {
    return __builtin_amdgcn_wmma_f32_16x16x32_bf16(false, a.v, false, b.v,
                                                   (short)0, c, false, false);
}

// ---------------------------------------------------------------------------
// K0: f32 -> bf16 weight conversion
// ---------------------------------------------------------------------------
__global__ void cvt_bf16_kernel(const float* __restrict__ src,
                                unsigned short* __restrict__ dst, int n) {
    int i = blockIdx.x * blockDim.x + threadIdx.x;
    if (i < n) dst[i] = f2bf(src[i]);
}

// ---------------------------------------------------------------------------
// K1: out = act(A @ W + b).  A: f32 [M,K], W: bf16 [K,N].  BM=64 BN=128 BK=32.
// 256 threads = 8 waves (wave grid 2x4, each wave a 32x32 = 2x2 frag tile).
// Double-buffered LDS: one barrier per K-step; global loads for tile t+1 are
// issued before the WMMAs of tile t so HBM/L2 latency overlaps matrix issue.
// ---------------------------------------------------------------------------
__global__ __launch_bounds__(256)
void gemm_bias_act_kernel(const float* __restrict__ A,
                          const unsigned short* __restrict__ W,
                          const float* __restrict__ bias,
                          float* __restrict__ out,
                          int M, int N, int K, int reluIn, int reluOut) {
    __shared__ unsigned short aT[2][64 * 34];
    __shared__ unsigned short bT[2][128 * 34];

    const int tid  = threadIdx.x;
    const int w    = tid >> 5;
    const int lane = tid & 31;
    const int wr   = w >> 2;          // 0..1
    const int wc   = w & 3;           // 0..3
    const int mb   = blockIdx.y * 64;
    const int nb   = blockIdx.x * 128;

    v8f acc[2][2];
#pragma unroll
    for (int i = 0; i < 2; ++i)
#pragma unroll
        for (int j = 0; j < 2; ++j)
            acc[i][j] = (v8f){0.f, 0.f, 0.f, 0.f, 0.f, 0.f, 0.f, 0.f};

    // A staging map: thread -> 8 consecutive floats
    const int aIdx = tid * 8;
    const int ar   = aIdx >> 5;
    const int ac   = aIdx & 31;
    // B staging map: thread -> 16 consecutive bf16 along n
    const int bIdx = tid * 16;
    const int br   = bIdx >> 7;
    const int bc   = bIdx & 127;

    float4         rA0, rA1;
    unsigned short rB[16];

    auto globalLoad = [&](int k0) {
        const float* ga = A + (size_t)(mb + ar) * K + k0 + ac;
        __builtin_prefetch(ga + 32, 0, 1);            // -> global_prefetch_b8
        rA0 = *(const float4*)ga;
        rA1 = *(const float4*)(ga + 4);
        const unsigned short* gb = W + (size_t)(k0 + br) * N + nb + bc;
#pragma unroll
        for (int i = 0; i < 16; ++i) rB[i] = gb[i];
    };
    auto ldsStore = [&](int buf) {
        float v[8] = {rA0.x, rA0.y, rA0.z, rA0.w, rA1.x, rA1.y, rA1.z, rA1.w};
        if (reluIn) {
#pragma unroll
            for (int i = 0; i < 8; ++i) v[i] = fmaxf(v[i], 0.f);
        }
        unsigned short* ad = aT[buf] + ar * 34 + ac;
#pragma unroll
        for (int i = 0; i < 8; ++i) ad[i] = f2bf(v[i]);
#pragma unroll
        for (int i = 0; i < 16; ++i)
            bT[buf][(bc + i) * 34 + br] = rB[i];
    };

    globalLoad(0);
    ldsStore(0);
    __syncthreads();

    const int T = K >> 5;
    for (int t = 0; t < T; ++t) {
        const int buf = t & 1;
        if (t + 1 < T) globalLoad((t + 1) << 5);      // hide latency under WMMAs

        Frag a0, a1, b0, b1;
        load_a_frag(aT[buf], 34, wr * 32,      lane, a0);
        load_a_frag(aT[buf], 34, wr * 32 + 16, lane, a1);
        load_b_frag(bT[buf], 34, wc * 32,      lane, b0);
        load_b_frag(bT[buf], 34, wc * 32 + 16, lane, b1);
        acc[0][0] = wmma_bf16(a0, b0, acc[0][0]);
        acc[0][1] = wmma_bf16(a0, b1, acc[0][1]);
        acc[1][0] = wmma_bf16(a1, b0, acc[1][0]);
        acc[1][1] = wmma_bf16(a1, b1, acc[1][1]);

        if (t + 1 < T) ldsStore((t + 1) & 1);         // other buffer: no pre-barrier
        __syncthreads();                              // single barrier per K-step
    }

    const int hi = lane >> 4;
#pragma unroll
    for (int fr = 0; fr < 2; ++fr) {
#pragma unroll
        for (int fc = 0; fc < 2; ++fc) {
            int gn  = nb + wc * 32 + fc * 16 + (lane & 15);
            float bv = bias ? bias[gn] : 0.f;
#pragma unroll
            for (int g = 0; g < 8; ++g) {
                int gm  = mb + wr * 32 + fr * 16 + hi * 8 + g;
                float v = acc[fr][fc][g] + bv;
                if (reluOut) v = fmaxf(v, 0.f);
                out[(size_t)gm * N + gn] = v;
            }
        }
    }
}

// ---------------------------------------------------------------------------
// K2: fused two-matrix fusion:
//   out = hid + relu(inp)@W1 + b1 + relu(hid)@W2 + b2       (N == K == H)
// Same double-buffered pipeline; the two passes share one software pipeline
// (2*T steps, source pointers switch at the halfway point).
// ---------------------------------------------------------------------------
__global__ __launch_bounds__(256)
void fusion_kernel(const float* __restrict__ inp,
                   const unsigned short* __restrict__ W1,
                   const float* __restrict__ b1,
                   const float* __restrict__ hid,
                   const unsigned short* __restrict__ W2,
                   const float* __restrict__ b2,
                   float* __restrict__ out, int M, int N, int K) {
    __shared__ unsigned short aT[2][64 * 34];
    __shared__ unsigned short bT[2][128 * 34];

    const int tid  = threadIdx.x;
    const int w    = tid >> 5;
    const int lane = tid & 31;
    const int wr   = w >> 2;
    const int wc   = w & 3;
    const int mb   = blockIdx.y * 64;
    const int nb   = blockIdx.x * 128;

    v8f acc[2][2];
#pragma unroll
    for (int i = 0; i < 2; ++i)
#pragma unroll
        for (int j = 0; j < 2; ++j)
            acc[i][j] = (v8f){0.f, 0.f, 0.f, 0.f, 0.f, 0.f, 0.f, 0.f};

    const int aIdx = tid * 8;
    const int ar   = aIdx >> 5;
    const int ac   = aIdx & 31;
    const int bIdx = tid * 16;
    const int br   = bIdx >> 7;
    const int bc   = bIdx & 127;

    const int T  = K >> 5;   // steps per pass
    const int TT = T * 2;

    float4         rA0, rA1;
    unsigned short rB[16];

    auto globalLoad = [&](int st) {
        const float*          Asrc = (st < T) ? inp : hid;
        const unsigned short* Wsrc = (st < T) ? W1  : W2;
        const int k0 = ((st < T) ? st : st - T) << 5;
        const float* ga = Asrc + (size_t)(mb + ar) * K + k0 + ac;
        __builtin_prefetch(ga + 32, 0, 1);
        rA0 = *(const float4*)ga;
        rA1 = *(const float4*)(ga + 4);
        const unsigned short* gb = Wsrc + (size_t)(k0 + br) * N + nb + bc;
#pragma unroll
        for (int i = 0; i < 16; ++i) rB[i] = gb[i];
    };
    auto ldsStore = [&](int buf) {
        unsigned short* ad = aT[buf] + ar * 34 + ac;
        ad[0] = f2bf(fmaxf(rA0.x, 0.f)); ad[1] = f2bf(fmaxf(rA0.y, 0.f));
        ad[2] = f2bf(fmaxf(rA0.z, 0.f)); ad[3] = f2bf(fmaxf(rA0.w, 0.f));
        ad[4] = f2bf(fmaxf(rA1.x, 0.f)); ad[5] = f2bf(fmaxf(rA1.y, 0.f));
        ad[6] = f2bf(fmaxf(rA1.z, 0.f)); ad[7] = f2bf(fmaxf(rA1.w, 0.f));
#pragma unroll
        for (int i = 0; i < 16; ++i)
            bT[buf][(bc + i) * 34 + br] = rB[i];
    };

    globalLoad(0);
    ldsStore(0);
    __syncthreads();

    for (int st = 0; st < TT; ++st) {
        const int buf = st & 1;
        if (st + 1 < TT) globalLoad(st + 1);

        Frag a0, a1, bb0, bb1;
        load_a_frag(aT[buf], 34, wr * 32,      lane, a0);
        load_a_frag(aT[buf], 34, wr * 32 + 16, lane, a1);
        load_b_frag(bT[buf], 34, wc * 32,      lane, bb0);
        load_b_frag(bT[buf], 34, wc * 32 + 16, lane, bb1);
        acc[0][0] = wmma_bf16(a0, bb0, acc[0][0]);
        acc[0][1] = wmma_bf16(a0, bb1, acc[0][1]);
        acc[1][0] = wmma_bf16(a1, bb0, acc[1][0]);
        acc[1][1] = wmma_bf16(a1, bb1, acc[1][1]);

        if (st + 1 < TT) ldsStore((st + 1) & 1);
        __syncthreads();
    }

    const int hi = lane >> 4;
#pragma unroll
    for (int fr = 0; fr < 2; ++fr) {
#pragma unroll
        for (int fc = 0; fc < 2; ++fc) {
            int gn   = nb + wc * 32 + fc * 16 + (lane & 15);
            float bv = b1[gn] + b2[gn];
#pragma unroll
            for (int g = 0; g < 8; ++g) {
                int gm = mb + wr * 32 + fr * 16 + hi * 8 + g;
                out[(size_t)gm * N + gn] =
                    acc[fr][fc][g] + bv + hid[(size_t)gm * N + gn];
            }
        }
    }
}

// ---------------------------------------------------------------------------
// K3: fused gate unit.  For 16 relations per block:
//   x = concat(U[iu], P[ip])  (2048) -> LayerNorm -> relu -> [16,2048]x[2048,128]
//   gate[r] = mean(sigmoid(. + bW))
// colU/colP: -1 => row index r; 0/1 => pairs[2r+col].
// Double-buffered: the LayerNorm VALU work for step t+1 overlaps the WMMA of t.
// ---------------------------------------------------------------------------
__global__ __launch_bounds__(256)
void gate_kernel(const float* __restrict__ U, int colU,
                 const float* __restrict__ P, int colP,
                 const int* __restrict__ pairs,
                 const float* __restrict__ ln_g, const float* __restrict__ ln_b,
                 const unsigned short* __restrict__ W,
                 const float* __restrict__ bW,
                 float* __restrict__ gates) {
    __shared__ unsigned short aT[2][16 * 34];
    __shared__ unsigned short bT[2][128 * 34];
    __shared__ float redS[16 * 17];
    __shared__ float redQ[16 * 17];
    __shared__ float meanS[16];
    __shared__ float invS[16];
    __shared__ float gsum[16];

    const int tid  = threadIdx.x;
    const int w    = tid >> 5;
    const int lane = tid & 31;
    const int rb   = blockIdx.x * 16;
    const int row  = tid >> 4;        // 16 threads per row
    const int sl   = tid & 15;
    const int r    = rb + row;

    const int iu = (colU < 0) ? r : pairs[2 * r + colU];
    const int ip = (colP < 0) ? r : pairs[2 * r + colP];
    const float* u = U + (size_t)iu * H_;
    const float* p = P + (size_t)ip * H_;

    // ---- LayerNorm statistics ----
    float s = 0.f, q = 0.f;
    for (int j = sl; j < 2 * H_; j += 16) {
        float x = (j < H_) ? u[j] : p[j - H_];
        s += x;
        q += x * x;
    }
    redS[row * 17 + sl] = s;
    redQ[row * 17 + sl] = q;
    __syncthreads();
    if (tid < 16) {
        float ts = 0.f, tq = 0.f;
#pragma unroll
        for (int i = 0; i < 16; ++i) { ts += redS[tid * 17 + i]; tq += redQ[tid * 17 + i]; }
        float m  = ts * (1.0f / (2 * H_));
        float vv = tq * (1.0f / (2 * H_)) - m * m;
        meanS[tid] = m;
        invS[tid]  = rsqrtf(vv + 1e-5f);
        gsum[tid]  = 0.f;
    }
    __syncthreads();

    // ---- pipelined GEMM with fused normalize/relu/bf16-convert ----
    v8f acc = (v8f){0.f, 0.f, 0.f, 0.f, 0.f, 0.f, 0.f, 0.f};
    const int kk   = (tid & 15) * 2;      // two consecutive x elements per thread
    const int bIdx = tid * 16;
    const int brr  = bIdx >> 7;
    const int bcc  = bIdx & 127;
    const float m  = meanS[row];
    const float iv = invS[row];

    float          x0, x1, lg0, lg1, lb0, lb1;
    unsigned short rB[16];

    auto stageLoad = [&](int t) {
        int j = (t << 5) + kk;
        const float* base = (j < H_) ? (u + j) : (p + j - H_);
        x0  = base[0];  x1  = base[1];
        lg0 = ln_g[j];  lg1 = ln_g[j + 1];
        lb0 = ln_b[j];  lb1 = ln_b[j + 1];
        const unsigned short* gb = W + (size_t)((t << 5) + brr) * GW_ + bcc;
        __builtin_prefetch(gb + (size_t)32 * GW_, 0, 1);
#pragma unroll
        for (int i = 0; i < 16; ++i) rB[i] = gb[i];
    };
    auto stageStore = [&](int buf) {
        float h0 = fmaxf((x0 - m) * iv * lg0 + lb0, 0.f);
        float h1 = fmaxf((x1 - m) * iv * lg1 + lb1, 0.f);
        aT[buf][row * 34 + kk]     = f2bf(h0);
        aT[buf][row * 34 + kk + 1] = f2bf(h1);
#pragma unroll
        for (int i = 0; i < 16; ++i)
            bT[buf][(bcc + i) * 34 + brr] = rB[i];
    };

    stageLoad(0);
    stageStore(0);
    __syncthreads();

    const int T = (2 * H_) >> 5;   // 64
    for (int t = 0; t < T; ++t) {
        const int buf = t & 1;
        if (t + 1 < T) stageLoad(t + 1);

        Frag af, bf;
        load_a_frag(aT[buf], 34, 0, lane, af);       // one shared 16-row A frag
        load_b_frag(bT[buf], 34, w * 16, lane, bf);  // wave w owns cols [16w,16w+16)
        acc = wmma_bf16(af, bf, acc);

        if (t + 1 < T) stageStore((t + 1) & 1);
        __syncthreads();
    }

    // ---- sigmoid + mean over the 128 outputs per row ----
    const int n  = (w << 4) + (lane & 15);
    const int hi = lane >> 4;
    const float bwv = bW[n];
#pragma unroll
    for (int g = 0; g < 8; ++g) {
        float sg = sigmoidf_(acc[g] + bwv);
        atomicAdd(&gsum[hi * 8 + g], sg);       // LDS ds_add_f32
    }
    __syncthreads();
    if (tid < 16)
        gates[rb + tid] = gsum[tid] * (1.0f / GW_);
}

// ---------------------------------------------------------------------------
// K4: segment-sum accumulate: sum[idx] += rel[r]*gate[r]; cnt[idx] += 1
// ---------------------------------------------------------------------------
__global__ void accum_msg_kernel(const float* __restrict__ rel,
                                 const float* __restrict__ gates,
                                 const int* __restrict__ pairs, int col,
                                 float* __restrict__ sum, float* __restrict__ cnt) {
    const int r   = blockIdx.x;
    const float g = gates[r];
    const int  i  = pairs[2 * r + col];
    const float* src = rel + (size_t)r * H_;
    float*       dst = sum + (size_t)i * H_;
    for (int c = threadIdx.x; c < H_; c += blockDim.x)
        unsafeAtomicAdd(&dst[c], src[c] * g);   // -> global_atomic_add_f32
    if (threadIdx.x == 0)
        unsafeAtomicAdd(&cnt[i], 1.0f);
}

// K5: fin = 0.5*(safe_mean(sum1,cnt1) + safe_mean(sum2,cnt2))
__global__ void combine_msg_kernel(const float* __restrict__ s1, const float* __restrict__ c1,
                                   const float* __restrict__ s2, const float* __restrict__ c2,
                                   float* __restrict__ fin, int total) {
    int i = blockIdx.x * blockDim.x + threadIdx.x;
    if (i >= total) return;
    int row = i >> 10;
    float n1 = c1[row], n2 = c2[row];
    float m1 = (n1 > 0.f) ? s1[i] / fmaxf(n1, 1.f) : 0.f;
    float m2 = (n2 > 0.f) ? s2[i] / fmaxf(n2, 1.f) : 0.f;
    fin[i] = 0.5f * (m1 + m2);
}

// K6: P[r] = 0.5*(inst[subj]*g3[r] + inst[obj]*g4[r])
__global__ void combine_pred_kernel(const float* __restrict__ inst,
                                    const float* __restrict__ g3,
                                    const float* __restrict__ g4,
                                    const int* __restrict__ pairs,
                                    float* __restrict__ Pout, int total) {
    int i = blockIdx.x * blockDim.x + threadIdx.x;
    if (i >= total) return;
    int r = i >> 10, c = i & (H_ - 1);
    int s = pairs[2 * r], o = pairs[2 * r + 1];
    Pout[i] = 0.5f * (inst[(size_t)s * H_ + c] * g3[r] +
                      inst[(size_t)o * H_ + c] * g4[r]);
}

// ---------------------------------------------------------------------------
// Host launcher
// ---------------------------------------------------------------------------
extern "C" void kernel_launch(void* const* d_in, const int* in_sizes, int n_in,
                              void* d_out, int out_size, void* d_ws, size_t ws_size,
                              hipStream_t stream) {
    (void)in_sizes; (void)n_in; (void)out_size; (void)ws_size;

    const float* inst_feat = (const float*)d_in[0];
    const float* rel_feat  = (const float*)d_in[1];
    const float* Wobj      = (const float*)d_in[2];
    const float* bObj      = (const float*)d_in[3];
    const float* Wrel      = (const float*)d_in[4];
    const float* bRel      = (const float*)d_in[5];
    // gate params: [6..9]=pred2sub, [10..13]=pred2obj, [14..17]=sub2pred, [18..21]=obj2pred
    const float* gate_lng[4] = {(const float*)d_in[6],  (const float*)d_in[10],
                                (const float*)d_in[14], (const float*)d_in[18]};
    const float* gate_lnb[4] = {(const float*)d_in[7],  (const float*)d_in[11],
                                (const float*)d_in[15], (const float*)d_in[19]};
    const float* gate_W[4]   = {(const float*)d_in[8],  (const float*)d_in[12],
                                (const float*)d_in[16], (const float*)d_in[20]};
    const float* gate_b[4]   = {(const float*)d_in[9],  (const float*)d_in[13],
                                (const float*)d_in[17], (const float*)d_in[21]};
    const float* of_wih = (const float*)d_in[22];
    const float* of_bih = (const float*)d_in[23];
    const float* of_whh = (const float*)d_in[24];
    const float* of_bhh = (const float*)d_in[25];
    const float* pf_wih = (const float*)d_in[26];
    const float* pf_bih = (const float*)d_in[27];
    const float* pf_whh = (const float*)d_in[28];
    const float* pf_bhh = (const float*)d_in[29];
    const int*   pairs  = (const int*)d_in[30];

    // ---- workspace carve-out ----
    char* ws = (char*)d_ws;
    size_t off = 0;
    auto alloc = [&](size_t bytes) -> char* {
        char* p = ws + off;
        off = (off + bytes + 255) & ~(size_t)255;
        return p;
    };
    unsigned short* wobj_b = (unsigned short*)alloc((size_t)POOL_ * H_ * 2);
    unsigned short* wrel_b = (unsigned short*)alloc((size_t)POOL_ * H_ * 2);
    unsigned short* wg_b[4];
    for (int i = 0; i < 4; ++i) wg_b[i] = (unsigned short*)alloc((size_t)POOL_ * GW_ * 2);
    unsigned short* of_wih_b = (unsigned short*)alloc((size_t)H_ * H_ * 2);
    unsigned short* of_whh_b = (unsigned short*)alloc((size_t)H_ * H_ * 2);
    unsigned short* pf_wih_b = (unsigned short*)alloc((size_t)H_ * H_ * 2);
    unsigned short* pf_whh_b = (unsigned short*)alloc((size_t)H_ * H_ * 2);
    float* instA = (float*)alloc((size_t)NI_ * H_ * 4);
    float* instB = (float*)alloc((size_t)NI_ * H_ * 4);
    float* relA  = (float*)alloc((size_t)NR_ * H_ * 4);
    float* relB  = (float*)alloc((size_t)NR_ * H_ * 4);
    float* fin   = (float*)alloc((size_t)NI_ * H_ * 4);
    float* Pbuf  = (float*)alloc((size_t)NR_ * H_ * 4);
    float* sum1  = (float*)alloc((size_t)NI_ * H_ * 4);
    float* sum2  = (float*)alloc((size_t)NI_ * H_ * 4);
    float* cnt1  = (float*)alloc((size_t)NI_ * 4);
    float* cnt2  = (float*)alloc((size_t)NI_ * 4);
    float* g1    = (float*)alloc((size_t)NR_ * 4);
    float* g2    = (float*)alloc((size_t)NR_ * 4);
    float* g3    = (float*)alloc((size_t)NR_ * 4);
    float* g4    = (float*)alloc((size_t)NR_ * 4);

    auto cvt = [&](const float* src, unsigned short* dst, size_t n) {
        cvt_bf16_kernel<<<(unsigned)((n + 255) / 256), 256, 0, stream>>>(src, dst, (int)n);
    };
    cvt(Wobj, wobj_b, (size_t)POOL_ * H_);
    cvt(Wrel, wrel_b, (size_t)POOL_ * H_);
    for (int i = 0; i < 4; ++i) cvt(gate_W[i], wg_b[i], (size_t)POOL_ * GW_);
    cvt(of_wih, of_wih_b, (size_t)H_ * H_);
    cvt(of_whh, of_whh_b, (size_t)H_ * H_);
    cvt(pf_wih, pf_wih_b, (size_t)H_ * H_);
    cvt(pf_whh, pf_whh_b, (size_t)H_ * H_);

    // ---- down-projections ----
    gemm_bias_act_kernel<<<dim3(H_ / 128, NI_ / 64), 256, 0, stream>>>(
        inst_feat, wobj_b, bObj, instA, NI_, H_, POOL_, 0, 1);
    gemm_bias_act_kernel<<<dim3(H_ / 128, NR_ / 64), 256, 0, stream>>>(
        rel_feat, wrel_b, bRel, relA, NR_, H_, POOL_, 0, 1);

    float* out_inst = (float*)d_out;
    float* out_rel  = (float*)d_out + (size_t)NI_ * H_;

    float* inst_cur = instA;
    float* rel_cur  = relA;
    for (int it = 0; it < 2; ++it) {
        float* inst_next = (it == 1) ? out_inst : instB;
        float* rel_next  = (it == 1) ? out_rel  : relB;

        // message gates: unary = inst[idx], pair = rel
        gate_kernel<<<NR_ / 16, 256, 0, stream>>>(
            inst_cur, 0, rel_cur, -1, pairs, gate_lng[0], gate_lnb[0], wg_b[0], gate_b[0], g1);
        gate_kernel<<<NR_ / 16, 256, 0, stream>>>(
            inst_cur, 1, rel_cur, -1, pairs, gate_lng[1], gate_lnb[1], wg_b[1], gate_b[1], g2);

        hipMemsetAsync(sum1, 0, (size_t)NI_ * H_ * 4, stream);
        hipMemsetAsync(sum2, 0, (size_t)NI_ * H_ * 4, stream);
        hipMemsetAsync(cnt1, 0, (size_t)NI_ * 4, stream);
        hipMemsetAsync(cnt2, 0, (size_t)NI_ * 4, stream);

        accum_msg_kernel<<<NR_, 256, 0, stream>>>(rel_cur, g1, pairs, 0, sum1, cnt1);
        accum_msg_kernel<<<NR_, 256, 0, stream>>>(rel_cur, g2, pairs, 1, sum2, cnt2);
        combine_msg_kernel<<<(NI_ * H_) / 256, 256, 0, stream>>>(
            sum1, cnt1, sum2, cnt2, fin, NI_ * H_);

        fusion_kernel<<<dim3(H_ / 128, NI_ / 64), 256, 0, stream>>>(
            fin, of_wih_b, of_bih, inst_cur, of_whh_b, of_bhh, inst_next, NI_, H_, H_);

        // pred gates: unary = rel, pair = inst[idx]
        gate_kernel<<<NR_ / 16, 256, 0, stream>>>(
            rel_cur, -1, inst_cur, 0, pairs, gate_lng[2], gate_lnb[2], wg_b[2], gate_b[2], g3);
        gate_kernel<<<NR_ / 16, 256, 0, stream>>>(
            rel_cur, -1, inst_cur, 1, pairs, gate_lng[3], gate_lnb[3], wg_b[3], gate_b[3], g4);
        combine_pred_kernel<<<((size_t)NR_ * H_) / 256, 256, 0, stream>>>(
            inst_cur, g3, g4, pairs, Pbuf, NR_ * H_);

        fusion_kernel<<<dim3(H_ / 128, NR_ / 64), 256, 0, stream>>>(
            Pbuf, pf_wih_b, pf_bih, rel_cur, pf_whh_b, pf_bhh, rel_next, NR_, H_, H_);

        inst_cur = inst_next;
        rel_cur  = rel_next;
    }
}
